// Discriminator_85641647882468
// MI455X (gfx1250) — matrix operand order repaired and checked by reading, not compiled, not run on hardware
//
#include <hip/hip_runtime.h>
#include <hip/hip_bf16.h>

// ---------- types ----------
typedef __attribute__((ext_vector_type(16))) _Float16 v16h;
typedef __attribute__((ext_vector_type(8)))  float    v8f;
typedef __attribute__((ext_vector_type(2)))  _Float16 h2;
typedef __attribute__((ext_vector_type(4)))  unsigned u32x4;
typedef __attribute__((ext_vector_type(4)))  int      i32x4;
typedef __attribute__((ext_vector_type(8)))  int      i32x8;

#define NROWS 4096
#define NB    64
#define NC    32
// padded dims (multiples of 32 for WMMA K, multiples of 16 for N)
#define K1 128   // x cols: 100 -> 128
#define N1 224   // h1 cols: 200 -> 224
#define N2 128   // h  cols: 100 -> 128
#define N3 2048  // Ms cols (already multiple of 32)

#define TJ 4     // j-rows per tile in minibatch kernel (2 x 16KB LDS buffers)

#if defined(__gfx1250__) && __has_builtin(__builtin_amdgcn_tensor_load_to_lds) && \
    __has_builtin(__builtin_amdgcn_s_wait_tensorcnt)
#define USE_TDM 1
#else
#define USE_TDM 0
#endif

__device__ inline h2 habs2_(h2 v) {
    unsigned u;
    __builtin_memcpy(&u, &v, 4);
    u &= 0x7FFF7FFFu;
    __builtin_memcpy(&v, &u, 4);
    return v;
}

#if USE_TDM
// Issue a TDM load of one contiguous 16KB j-tile (4096 dwords) into LDS.
// D# group0: count=1 | lds_addr | global_addr[56:0] | type=2 ("image")
// D# group1: data_size=4B, 1-D tile: tile_dim0 = tensor_dim0 = stride = 4096 dwords.
__device__ inline void tdm_load_tile(const void* gsrc, unsigned lds_base) {
    unsigned long long ga = (unsigned long long)(size_t)gsrc;
    u32x4 g0 = { 1u, lds_base, (unsigned)(ga & 0xFFFFFFFFu),
                 (unsigned)((ga >> 32) & 0x1FFFFFFu) | (2u << 30) };
    i32x8 g1 = { (int)(2u << 16),          // data_size=2 (4 bytes)
                 (int)(4096u << 16),       // tensor_dim0[15:0] << 16
                 (int)(1u << 16),          // tensor_dim1 = 1
                 (int)(4096u << 16),       // tile_dim0 = 4096 dwords
                 1,                        // tile_dim1 = 1
                 4096,                     // tensor_dim0_stride lo
                 0, 0 };
    i32x4 z4 = { 0, 0, 0, 0 };
#if __clang_major__ >= 23
    i32x8 z8 = { 0, 0, 0, 0, 0, 0, 0, 0 };
    __builtin_amdgcn_tensor_load_to_lds(g0, g1, z4, z4, z8, 0);
#else
    __builtin_amdgcn_tensor_load_to_lds(g0, g1, z4, z4, 0);
#endif
}
#endif

// ---------- pack / pad kernels (compile-time dims -> no runtime div/mod) ----------
// A-matrix pack (row-major, f32->f16, zero pad)
template <int RIN, int CIN, int ROUT, int COUT>
__global__ void pack_a(const float* __restrict__ src, _Float16* __restrict__ dst) {
    int idx = blockIdx.x * 256 + threadIdx.x;
    if (idx >= ROUT * COUT) return;
    int r = idx / COUT, c = idx % COUT;
    dst[idx] = (r < RIN && c < CIN) ? (_Float16)src[r * CIN + c] : (_Float16)0.f;
}

// B-matrix pack, FRAGMENT-MAJOR: for each 32x16 (k,n) tile, 32 lanes x 8 dwords contiguous,
// exactly the WMMA B-fragment register image -> the GEMM loads it as two b128 per wave.
// Lane l covers column n = nt*16 + (l&15), K half (l>=16 ? +16 : 0); dword v holds {k, k+1}.
// pack_bt_frag: logical B[k][n] = W[n][k] (transposed weights).
template <int KIN, int NIN, int KOUT, int NOUT>
__global__ void pack_bt_frag(const float* __restrict__ W, _Float16* __restrict__ dst) {
    int o = blockIdx.x * 256 + threadIdx.x;  // flat f16 index
    if (o >= KOUT * NOUT) return;
    int hi = o & 1, dw = o >> 1;
    int v = dw & 7, lane = (dw >> 3) & 31, tile = dw >> 8;
    const int NT = NOUT >> 4;
    int nt = tile % NT, kt = tile / NT;
    int n = nt * 16 + (lane & 15);
    int k = kt * 32 + ((lane >> 4) << 4) + 2 * v + hi;
    dst[o] = (k < KIN && n < NIN) ? (_Float16)W[n * KIN + k] : (_Float16)0.f;
}

// pack_b_frag: logical B[k][n] = src[k][n] (no transpose; for T).
template <int KIN, int NIN, int KOUT, int NOUT>
__global__ void pack_b_frag(const float* __restrict__ src, _Float16* __restrict__ dst) {
    int o = blockIdx.x * 256 + threadIdx.x;
    if (o >= KOUT * NOUT) return;
    int hi = o & 1, dw = o >> 1;
    int v = dw & 7, lane = (dw >> 3) & 31, tile = dw >> 8;
    const int NT = NOUT >> 4;
    int nt = tile % NT, kt = tile / NT;
    int n = nt * 16 + (lane & 15);
    int k = kt * 32 + ((lane >> 4) << 4) + 2 * v + hi;
    dst[o] = (k < KIN && n < NIN) ? (_Float16)src[k * NIN + n] : (_Float16)0.f;
}

__global__ void pack_bias(const float* __restrict__ b, float* __restrict__ dst,
                          int nin, int nout) {
    int idx = blockIdx.x * 256 + threadIdx.x;
    if (idx >= nout) return;
    dst[idx] = (idx < nin) ? b[idx] : 0.f;
}

// ---------- WMMA GEMM:  C[MxN] = act(A16[MxK] @ Bfrag[KxN] + bias) ----------
// Block = 256 threads = 8 waves. blockIdx.x -> 16-row m-tile, blockIdx.y*8+wave -> 16-col n-tile.
// A tile staged in LDS (two ds_load_b128 per fragment); B is fragment-major (two global b128).
template <bool RELU, bool BIAS, bool OUTF, bool OUTH>
__global__ __launch_bounds__(256)
void wmma_gemm(const _Float16* __restrict__ A, const _Float16* __restrict__ B,
               const float* __restrict__ bias, float* __restrict__ Cf,
               _Float16* __restrict__ Ch, int M, int N, int K) {
    __shared__ __align__(16) _Float16 sA[16 * 224];  // max K = 224

    int m0 = blockIdx.x * 16;
    const _Float16* Ag = A + (size_t)m0 * K;
    for (int idx = threadIdx.x; idx < 16 * K; idx += 256) sA[idx] = Ag[idx];
    __syncthreads();

    int wave = threadIdx.x >> 5;
    int lane = threadIdx.x & 31;
    int n0 = (blockIdx.y * 8 + wave) * 16;
    if (n0 >= N) return;

    int m  = lane & 15;
    int ka = (lane < 16) ? 0 : 16;     // A-frag K sub-base per lane half (halfs)
    int nn = n0 + (lane & 15);         // C column owned by this lane
    int ntiles = N >> 4;

    const uint4* Bf = (const uint4*)B;

    v8f acc = {};
    for (int k0 = 0; k0 < K; k0 += 32) {
        v16h a, b;
        // A fragment: two contiguous 16B chunks per lane -> ds_load_b128 x2
        const _Float16* arow = sA + m * K + k0 + ka;
        uint4* ap4 = (uint4*)&a;
        ap4[0] = *(const uint4*)(arow);
        ap4[1] = *(const uint4*)(arow + 8);
        // B fragment: 32B contiguous per lane in fragment-major layout -> global b128 x2
        const uint4* bt = Bf + ((size_t)((k0 >> 5) * ntiles + (n0 >> 4)) * 32 + lane) * 2;
        uint4* bp4 = (uint4*)&b;
        bp4[0] = bt[0];
        bp4[1] = bt[1];

        acc = __builtin_amdgcn_wmma_f32_16x16x32_f16(
            /*neg_a=*/false, a, /*neg_b=*/false, b,
            /*c_mod=*/(short)0, acc, /*reuse_a=*/false, /*reuse_b=*/false);
    }

    // C/D layout: VGPR v -> row m0 + v + (lane<16 ? 0 : 8), col nn
    float bv = BIAS ? bias[nn] : 0.f;
    int mb = m0 + ((lane < 16) ? 0 : 8);
    #pragma unroll
    for (int v = 0; v < 8; ++v) {
        float x = acc[v] + bv;
        if (RELU) x = fmaxf(x, 0.f);
        size_t off = (size_t)(mb + v) * N + nn;
        if (OUTF) Cf[off] = x;
        if (OUTH) Ch[off] = (_Float16)x;
    }
}

// ---------- minibatch discrimination: outT[i,b] = sum_j exp(-||Ms[i,b,:]-Ms[j,b,:]||_1) ----------
// 256 threads: thread t owns row i = blk*16 + (t&15) and 4 kernels b = (t>>4)*4 .. +3.
// Ms[i] slice (4 b x 32 c = 64 half2) cached in registers. Ms[j] tiles (TJ rows, 16KB) are
// DOUBLE-BUFFERED in LDS: wave 0 waits TENSORcnt for the current tile, immediately issues the
// TDM load of the next tile into the other buffer, so the DMA overlaps the whole compute step.
__global__ __launch_bounds__(256)
void minibatch_l1(const _Float16* __restrict__ Ms, float* __restrict__ outT, int nrows) {
    __shared__ __align__(16) h2 sj[2][TJ * 1024];  // 2 x (TJ rows x 2048 halfs) = 32 KB

    int t  = threadIdx.x;
    int il = t & 15;
    int bg = t >> 4;  // 0..15, each covers 4 of the 64 kernels
    int i  = blockIdx.x * 16 + il;

    const h2* mrow = (const h2*)(Ms + (size_t)i * N3) + bg * 64;
    h2 mi[64];
    #pragma unroll
    for (int c = 0; c < 64; ++c) mi[c] = mrow[c];

    float acc[4] = {0.f, 0.f, 0.f, 0.f};
    int ntj = nrows / TJ;

#if USE_TDM
    if (t < 32)  // wave 0 drives the TDM (EXEC is ignored by tensor ops)
        tdm_load_tile(Ms, (unsigned)(size_t)(void*)&sj[0][0]);
#else
    {  // prologue: fill buffer 0
        const uint4* s4 = (const uint4*)Ms;
        uint4* d4 = (uint4*)&sj[0][0];
        #pragma unroll
        for (int l = 0; l < TJ; ++l) d4[t + l * 256] = s4[t + l * 256];
    }
#endif

    for (int jt = 0; jt < ntj; ++jt) {
        int cur = jt & 1;
#if USE_TDM
        if (t < 32) {
            __builtin_amdgcn_s_wait_tensorcnt(0);     // current tile landed
            if (jt + 1 < ntj)                          // kick next tile -> other buffer
                tdm_load_tile(Ms + (size_t)(jt + 1) * TJ * N3,
                              (unsigned)(size_t)(void*)&sj[cur ^ 1][0]);
        }
#else
        if (jt + 1 < ntj) {
            const uint4* s4 = (const uint4*)(Ms + (size_t)(jt + 1) * TJ * N3);
            uint4* d4 = (uint4*)&sj[cur ^ 1][0];
            #pragma unroll
            for (int l = 0; l < TJ; ++l) d4[t + l * 256] = s4[t + l * 256];
            __builtin_prefetch(Ms + (size_t)(jt + 2) * TJ * N3 + t * 64, 0, 0);
        }
#endif
        __syncthreads();

        for (int jl = 0; jl < TJ; ++jl) {
            const h2* r = &sj[cur][jl * 1024 + bg * 64];
            #pragma unroll
            for (int b = 0; b < 4; ++b) {
                h2 s = (h2)(_Float16)0.f;
                #pragma unroll
                for (int c = 0; c < 16; ++c) {
                    h2 d = mi[b * 16 + c] - r[b * 16 + c];  // v_pk f16 math
                    s += habs2_(d);
                }
                float l1 = (float)s.x + (float)s.y;
                acc[b] += __expf(-l1);
            }
        }
        __syncthreads();  // everyone done reading sj[cur] before it is refilled
    }

    float* o = outT + (size_t)i * NB + bg * 4;
    #pragma unroll
    for (int b = 0; b < 4; ++b) o[b] = acc[b];
}

// ---------- final head: sigmoid(concat(h[:100], outT) @ W3.T + b3) ----------
__global__ __launch_bounds__(256)
void final_head(const float* __restrict__ h, const float* __restrict__ outT,
                const float* __restrict__ W3, const float* __restrict__ b3,
                float* __restrict__ out, int nrows) {
    int gid  = blockIdx.x * 256 + threadIdx.x;
    int row  = gid >> 5;
    int lane = threadIdx.x & 31;
    if (row >= nrows) return;
    float s = 0.f;
    for (int c = lane; c < 100; c += 32) s += h[(size_t)row * N2 + c] * W3[c];
    for (int b = lane; b < NB; b += 32) s += outT[(size_t)row * NB + b] * W3[100 + b];
    #pragma unroll
    for (int off = 16; off > 0; off >>= 1) s += __shfl_down(s, off, 32);
    if (lane == 0) out[row] = 1.f / (1.f + __expf(-(s + b3[0])));
}

// ---------- host launcher ----------
extern "C" void kernel_launch(void* const* d_in, const int* in_sizes, int n_in,
                              void* d_out, int out_size, void* d_ws, size_t ws_size,
                              hipStream_t stream) {
    const float* x  = (const float*)d_in[0];  // [4096,100]
    const float* W1 = (const float*)d_in[1];  // [200,100]
    const float* b1 = (const float*)d_in[2];  // [200]
    const float* W2 = (const float*)d_in[3];  // [100,200]
    const float* b2 = (const float*)d_in[4];  // [100]
    const float* T  = (const float*)d_in[5];  // [100,2048]
    const float* W3 = (const float*)d_in[6];  // [1,164]
    const float* b3 = (const float*)d_in[7];  // [1]
    float* out = (float*)d_out;

    char* ws = (char*)d_ws;
    auto alloc = [&](size_t bytes) {
        char* p = ws;
        ws += (bytes + 255) & ~(size_t)255;
        return p;
    };
    _Float16* x16  = (_Float16*)alloc((size_t)NROWS * K1 * 2);
    _Float16* w1t  = (_Float16*)alloc((size_t)K1 * N1 * 2);
    float*    b1p  = (float*)   alloc((size_t)N1 * 4);
    _Float16* h1   = (_Float16*)alloc((size_t)NROWS * N1 * 2);
    _Float16* w2t  = (_Float16*)alloc((size_t)N1 * N2 * 2);
    float*    b2p  = (float*)   alloc((size_t)N2 * 4);
    float*    hf   = (float*)   alloc((size_t)NROWS * N2 * 4);
    _Float16* h16  = (_Float16*)alloc((size_t)NROWS * N2 * 2);
    _Float16* T16  = (_Float16*)alloc((size_t)K1 * N3 * 2);
    _Float16* Ms   = (_Float16*)alloc((size_t)NROWS * N3 * 2);
    float*    oT   = (float*)   alloc((size_t)NROWS * NB * 4);

    auto blocks = [](int n) { return (n + 255) / 256; };

    // pack/pad inputs to f16 (weights in fragment-major WMMA-B layout)
    pack_a<NROWS, 100, NROWS, K1><<<blocks(NROWS * K1), 256, 0, stream>>>(x, x16);
    pack_bt_frag<100, 200, K1, N1><<<blocks(K1 * N1), 256, 0, stream>>>(W1, w1t);
    pack_bias<<<blocks(N1), 256, 0, stream>>>(b1, b1p, 200, N1);
    pack_bt_frag<200, 100, N1, N2><<<blocks(N1 * N2), 256, 0, stream>>>(W2, w2t);
    pack_bias<<<blocks(N2), 256, 0, stream>>>(b2, b2p, 100, N2);
    pack_b_frag<100, N3, K1, N3><<<blocks(K1 * N3), 256, 0, stream>>>(T, T16);

    // h1 = relu(x @ W1.T + b1)        [4096 x 224] f16
    wmma_gemm<true, true, false, true>
        <<<dim3(NROWS / 16, (N1 + 127) / 128), 256, 0, stream>>>(
            x16, w1t, b1p, nullptr, h1, NROWS, N1, K1);
    // h = relu(h1 @ W2.T + b2)        [4096 x 128] (f32 + f16 copies)
    wmma_gemm<true, true, true, true>
        <<<dim3(NROWS / 16, (N2 + 127) / 128), 256, 0, stream>>>(
            h1, w2t, b2p, hf, h16, NROWS, N2, N1);
    // Ms = h @ T                      [4096 x 2048] f16
    wmma_gemm<false, false, false, true>
        <<<dim3(NROWS / 16, N3 / 128), 256, 0, stream>>>(
            h16, T16, nullptr, nullptr, Ms, NROWS, N3, K1);

    // outT[i,b] = sum_j exp(-L1)      [4096 x 64]
    minibatch_l1<<<NROWS / 16, 256, 0, stream>>>(Ms, oT, NROWS);

    // out = sigmoid([h, outT] @ W3.T + b3)
    final_head<<<blocks(NROWS * 32), 256, 0, stream>>>(hf, oT, W3, b3, out, NROWS);

    (void)in_sizes; (void)n_in; (void)out_size; (void)ws_size;
}
